// Agent_46505905881806
// MI455X (gfx1250) — matrix operand order repaired and checked by reading, not compiled, not run on hardware
//
#include <hip/hip_runtime.h>
#include <hip/hip_bf16.h>
#include <math.h>

#define OBS 128
#define Hd  128
#define F1d 256
#define Ad  18
#define Td  128
#define Bd  1024
#define Nn  (Td * Bd)      // 131072 rows
#define G4  (4 * Hd)       // 512 gate cols

typedef __attribute__((ext_vector_type(16))) __bf16 v16bf;
typedef __attribute__((ext_vector_type(8)))  float  v8f;

union FragU { v16bf v; unsigned int d[8]; };

__device__ __forceinline__ unsigned short f2bf(float f) {
  unsigned int u = __float_as_uint(f);
  u = (u + 0x7FFFu + ((u >> 16) & 1u)) >> 16;
  return (unsigned short)u;
}

__device__ __forceinline__ float sigm(float x) { return 1.f / (1.f + __expf(-x)); }

// A fragment (16x32 bf16): lane m=lane&15; lanes 0-15 hold K {0..15},
// lanes 16-31 hold K {16..31}. Source row-major [rows][ld] bf16.
__device__ __forceinline__ v16bf load_frag_a(const unsigned short* base, int row0,
                                             int ld, int kbase, int lane) {
  const int m = lane & 15;
  const int khalf = (lane >> 4) << 4;
  const unsigned int* p = (const unsigned int*)(base + (size_t)(row0 + m) * ld);
  FragU f;
#pragma unroll
  for (int v = 0; v < 8; ++v) f.d[v] = p[(kbase + khalf + 2 * v) >> 1];
  return f.v;
}

// B fragment (32x16 bf16, B[k][n] = w[n][k]): lane n=lane&15; VGPR v holds K pair
// (2v,2v+1) for lanes 0-15, (16+2v,17+2v) for lanes 16-31. w row-major [outN][ldK].
__device__ __forceinline__ v16bf load_frag_b(const unsigned short* w, int n0,
                                             int ld, int kbase, int lane) {
  const int n = lane & 15;
  const int koff = (lane >> 4) << 4;
  const unsigned int* p = (const unsigned int*)(w + (size_t)(n0 + n) * ld);
  FragU f;
#pragma unroll
  for (int v = 0; v < 8; ++v) f.d[v] = p[(kbase + koff + 2 * v) >> 1];
  return f.v;
}

__device__ __forceinline__ v8f wmma_bf(v16bf a, v16bf b, v8f c) {
  return __builtin_amdgcn_wmma_f32_16x16x32_bf16(false, a, false, b, (short)0, c,
                                                 false, false);
}

// ---------------- weight fp32 -> bf16 ----------------
__global__ void cvt_bf16_kernel(const float* __restrict__ src,
                                unsigned short* __restrict__ dst, int n) {
  int i = blockIdx.x * blockDim.x + threadIdx.x;
  if (i < n) dst[i] = f2bf(src[i]);
}

// ---------------- 4-way f32 copy (state init / final state out) ----------------
struct Copy4 { const float* s[4]; float* d[4]; int n; };
__global__ void copy4_kernel(Copy4 p) {
  int i = blockIdx.x * blockDim.x + threadIdx.x;
  if (i >= p.n) return;
#pragma unroll
  for (int j = 0; j < 4; ++j) p.d[j][i] = p.s[j][i];
}

// ---------------- fused feature MLP: relu(LN(relu(x@w1^T+b1)@w2^T+b2)) ----------------
struct FeatParams {
  const float* x;
  const unsigned short* w1[2];
  const float* b1[2];
  const unsigned short* w2[2];
  const float* b2[2];
  const float* lnw[2];
  const float* lnb[2];
  unsigned short* feat[2];
};

__global__ __launch_bounds__(256) void feat_kernel(FeatParams p) {
  const int br = blockIdx.y;
  const int rows0 = blockIdx.x * 64;
  const int tid = threadIdx.x;
  const int wid = tid >> 5;
  const int lane = tid & 31;
  const int hi = lane >> 4;

  __shared__ __align__(16) unsigned char smem[49152];
  unsigned short* xt = (unsigned short*)smem;            // [64][128] bf16 (phase 1)
  unsigned short* h1 = (unsigned short*)(smem + 16384);  // [64][256] bf16
  float* h2 = (float*)smem;                              // [64][128] f32 (reuses xt+h1lo)

  // x tile -> bf16 LDS
#pragma unroll
  for (int i = 0; i < 32; ++i) {
    int e = tid + 256 * i;
    int r = e >> 7, k = e & 127;
    xt[e] = f2bf(p.x[(size_t)(rows0 + r) * OBS + k]);
  }
  __syncthreads();

  const v8f vz = {0.f, 0.f, 0.f, 0.f, 0.f, 0.f, 0.f, 0.f};

  // GEMM1: [64x128] x [128x256]; wave wid -> col tiles {2w, 2w+1}
  {
    v8f acc[4][2];
#pragma unroll
    for (int rt = 0; rt < 4; ++rt) { acc[rt][0] = vz; acc[rt][1] = vz; }
#pragma unroll
    for (int kb = 0; kb < 128; kb += 32) {
      v16bf a[4];
#pragma unroll
      for (int rt = 0; rt < 4; ++rt) a[rt] = load_frag_a(xt, rt * 16, OBS, kb, lane);
#pragma unroll
      for (int j = 0; j < 2; ++j) {
        v16bf b = load_frag_b(p.w1[br], (2 * wid + j) * 16, OBS, kb, lane);
#pragma unroll
        for (int rt = 0; rt < 4; ++rt) acc[rt][j] = wmma_bf(a[rt], b, acc[rt][j]);
      }
    }
#pragma unroll
    for (int j = 0; j < 2; ++j) {
      int col = (2 * wid + j) * 16 + (lane & 15);
      float bv = p.b1[br][col];
#pragma unroll
      for (int rt = 0; rt < 4; ++rt)
#pragma unroll
        for (int r = 0; r < 8; ++r) {
          int row = rt * 16 + r + 8 * hi;
          float v = acc[rt][j][r] + bv;
          h1[row * F1d + col] = f2bf(fmaxf(v, 0.f));
        }
    }
  }
  __syncthreads();

  // GEMM2: [64x256] x [256x128]; wave wid -> col tile wid
  {
    v8f acc[4];
#pragma unroll
    for (int rt = 0; rt < 4; ++rt) acc[rt] = vz;
#pragma unroll
    for (int kb = 0; kb < 256; kb += 32) {
      v16bf b = load_frag_b(p.w2[br], wid * 16, F1d, kb, lane);
#pragma unroll
      for (int rt = 0; rt < 4; ++rt) {
        v16bf a = load_frag_a(h1, rt * 16, F1d, kb, lane);
        acc[rt] = wmma_bf(a, b, acc[rt]);
      }
    }
    __syncthreads();  // h1 fully consumed; region 0..32KB reusable
    int col = wid * 16 + (lane & 15);
    float bv = p.b2[br][col];
#pragma unroll
    for (int rt = 0; rt < 4; ++rt)
#pragma unroll
      for (int r = 0; r < 8; ++r) {
        int row = rt * 16 + r + 8 * hi;
        h2[row * Hd + col] = acc[rt][r] + bv;
      }
  }
  __syncthreads();

  // LayerNorm + relu -> feat bf16
  if (tid < 64) {
    const float* row = h2 + tid * Hd;
    float s = 0.f;
    for (int k = 0; k < Hd; ++k) s += row[k];
    float mu = s * (1.f / Hd);
    float var = 0.f;
    for (int k = 0; k < Hd; ++k) { float d = row[k] - mu; var += d * d; }
    float rs = rsqrtf(var * (1.f / Hd) + 1e-5f);
    unsigned short* out = p.feat[br] + (size_t)(rows0 + tid) * Hd;
    for (int k = 0; k < Hd; ++k) {
      float y = (row[k] - mu) * rs * p.lnw[br][k] + p.lnb[br][k];
      out[k] = f2bf(fmaxf(y, 0.f));
    }
  }
}

// ---------------- one masked-LSTM timestep ----------------
struct StepParams {
  const int* done;
  const unsigned short* feat[2];
  const unsigned short* wih[2];
  const unsigned short* whh[2];
  const float* bias[2];
  float* h[2];
  float* c[2];
  float* hs[2];
  int t;
};

__global__ __launch_bounds__(256) void lstm_step_kernel(StepParams p) {
  const int br = blockIdx.z;
  const int rows0 = blockIdx.x * 32;
  const int tid = threadIdx.x;
  const int wid = tid >> 5;
  const int lane = tid & 31;
  const int hi = lane >> 4;
  const int t = p.t;

  __shared__ unsigned short htile[32 * Hd];  // masked h, bf16

#pragma unroll
  for (int i = 0; i < 16; ++i) {
    int e = tid + 256 * i;
    int r = e >> 7, k = e & 127;
    int brow = rows0 + r;
    float m = 1.f - (float)p.done[t * Bd + brow];
    htile[e] = f2bf(p.h[br][(size_t)brow * Hd + k] * m);
  }
  __syncthreads();

  const v8f vz = {0.f, 0.f, 0.f, 0.f, 0.f, 0.f, 0.f, 0.f};
  v8f acc[2][4];  // [row tile][gate]; wave wid covers gate cols g*128 + wid*16 ..+15
#pragma unroll
  for (int rt = 0; rt < 2; ++rt)
#pragma unroll
    for (int g = 0; g < 4; ++g) acc[rt][g] = vz;

  // input projection: feat_t @ wih^T  (feat already bf16 in ws)
  const unsigned short* featb = p.feat[br] + (size_t)(t * Bd + rows0) * Hd;
#pragma unroll
  for (int kb = 0; kb < 128; kb += 32) {
    v16bf a0 = load_frag_a(featb, 0, Hd, kb, lane);
    v16bf a1 = load_frag_a(featb, 16, Hd, kb, lane);
#pragma unroll
    for (int g = 0; g < 4; ++g) {
      v16bf b = load_frag_b(p.wih[br], g * Hd + wid * 16, Hd, kb, lane);
      acc[0][g] = wmma_bf(a0, b, acc[0][g]);
      acc[1][g] = wmma_bf(a1, b, acc[1][g]);
    }
  }
  // recurrent projection: (h * mask) @ whh^T
#pragma unroll
  for (int kb = 0; kb < 128; kb += 32) {
    v16bf a0 = load_frag_a(htile, 0, Hd, kb, lane);
    v16bf a1 = load_frag_a(htile, 16, Hd, kb, lane);
#pragma unroll
    for (int g = 0; g < 4; ++g) {
      v16bf b = load_frag_b(p.whh[br], g * Hd + wid * 16, Hd, kb, lane);
      acc[0][g] = wmma_bf(a0, b, acc[0][g]);
      acc[1][g] = wmma_bf(a1, b, acc[1][g]);
    }
  }

  // gate math: i/f/g/o for (row, colh) all live in this lane's registers
  const int colh = wid * 16 + (lane & 15);
  const float bi = p.bias[br][0 * Hd + colh];
  const float bf_ = p.bias[br][1 * Hd + colh];
  const float bg = p.bias[br][2 * Hd + colh];
  const float bo = p.bias[br][3 * Hd + colh];
#pragma unroll
  for (int rt = 0; rt < 2; ++rt)
#pragma unroll
    for (int r = 0; r < 8; ++r) {
      int brow = rows0 + rt * 16 + r + 8 * hi;
      float iv = acc[rt][0][r] + bi;
      float fv = acc[rt][1][r] + bf_;
      float gv = acc[rt][2][r] + bg;
      float ov = acc[rt][3][r] + bo;
      float m = 1.f - (float)p.done[t * Bd + brow];
      size_t idx = (size_t)brow * Hd + colh;
      float cp = p.c[br][idx] * m;
      float cn = sigm(fv) * cp + sigm(iv) * tanhf(gv);
      float hn = sigm(ov) * tanhf(cn);
      p.c[br][idx] = cn;
      p.h[br][idx] = hn;
      p.hs[br][(size_t)(t * Bd + brow) * Hd + colh] = hn;
    }
}

// ---------------- actor head: LN -> logits -> log_softmax -> logp/entropy ----------------
struct ActorParams {
  const float* hs;
  const float* lnw;
  const float* lnb;
  const float* hw;   // [A][H]
  const float* hb;   // [A]
  const long long* action;
  float* out_action;
  float* out_logp;
  float* out_ent;
};

__global__ __launch_bounds__(256) void actor_head_kernel(ActorParams p) {
  const int wid = threadIdx.x >> 5;
  const int lane = threadIdx.x & 31;
  const int n = blockIdx.x * 8 + wid;

  __shared__ float ah[8][Hd];
  __shared__ float lg[8][24];

  const float* row = p.hs + (size_t)n * Hd;
  float x0 = row[lane], x1 = row[lane + 32], x2 = row[lane + 64], x3 = row[lane + 96];
  float s = x0 + x1 + x2 + x3;
#pragma unroll
  for (int o = 16; o; o >>= 1) s += __shfl_xor(s, o, 32);
  float mu = s * (1.f / Hd);
  float d0 = x0 - mu, d1 = x1 - mu, d2 = x2 - mu, d3 = x3 - mu;
  float v = d0 * d0 + d1 * d1 + d2 * d2 + d3 * d3;
#pragma unroll
  for (int o = 16; o; o >>= 1) v += __shfl_xor(v, o, 32);
  float rs = rsqrtf(v * (1.f / Hd) + 1e-5f);
  ah[wid][lane]      = d0 * rs * p.lnw[lane]      + p.lnb[lane];
  ah[wid][lane + 32] = d1 * rs * p.lnw[lane + 32] + p.lnb[lane + 32];
  ah[wid][lane + 64] = d2 * rs * p.lnw[lane + 64] + p.lnb[lane + 64];
  ah[wid][lane + 96] = d3 * rs * p.lnw[lane + 96] + p.lnb[lane + 96];
  // wave-internal LDS RAW: in-order per wave, no barrier needed
  if (lane < Ad) {
    float acc = p.hb[lane];
    const float* w = p.hw + lane * Hd;
    for (int k = 0; k < Hd; ++k) acc += ah[wid][k] * w[k];
    lg[wid][lane] = acc;
  }
  if (lane == 0) {
    float mx = -3.4e38f;
    for (int a = 0; a < Ad; ++a) mx = fmaxf(mx, lg[wid][a]);
    float se = 0.f;
    for (int a = 0; a < Ad; ++a) se += __expf(lg[wid][a] - mx);
    float lse = mx + __logf(se);
    float ent = 0.f;
    for (int a = 0; a < Ad; ++a) {
      float lp = lg[wid][a] - lse;
      ent -= __expf(lp) * lp;
    }
    long long act = p.action[n];
    p.out_action[n] = (float)act;
    p.out_logp[n] = lg[wid][(int)act] - lse;
    p.out_ent[n] = ent;
  }
}

// ---------------- critic head: LN -> scalar value ----------------
__global__ __launch_bounds__(256) void critic_head_kernel(const float* __restrict__ hs,
                                                          const float* __restrict__ lnw,
                                                          const float* __restrict__ lnb,
                                                          const float* __restrict__ hw,
                                                          const float* __restrict__ hb,
                                                          float* __restrict__ out) {
  const int wid = threadIdx.x >> 5;
  const int lane = threadIdx.x & 31;
  const int n = blockIdx.x * 8 + wid;
  const float* row = hs + (size_t)n * Hd;
  float x0 = row[lane], x1 = row[lane + 32], x2 = row[lane + 64], x3 = row[lane + 96];
  float s = x0 + x1 + x2 + x3;
#pragma unroll
  for (int o = 16; o; o >>= 1) s += __shfl_xor(s, o, 32);
  float mu = s * (1.f / Hd);
  float d0 = x0 - mu, d1 = x1 - mu, d2 = x2 - mu, d3 = x3 - mu;
  float v = d0 * d0 + d1 * d1 + d2 * d2 + d3 * d3;
#pragma unroll
  for (int o = 16; o; o >>= 1) v += __shfl_xor(v, o, 32);
  float rs = rsqrtf(v * (1.f / Hd) + 1e-5f);
  float p0 = (d0 * rs * lnw[lane]      + lnb[lane])      * hw[lane];
  float p1 = (d1 * rs * lnw[lane + 32] + lnb[lane + 32]) * hw[lane + 32];
  float p2 = (d2 * rs * lnw[lane + 64] + lnb[lane + 64]) * hw[lane + 64];
  float p3 = (d3 * rs * lnw[lane + 96] + lnb[lane + 96]) * hw[lane + 96];
  float dot = p0 + p1 + p2 + p3;
#pragma unroll
  for (int o = 16; o; o >>= 1) dot += __shfl_xor(dot, o, 32);
  if (lane == 0) out[n] = dot + hb[0];
}

// ---------------- host orchestration ----------------
extern "C" void kernel_launch(void* const* d_in, const int* in_sizes, int n_in,
                              void* d_out, int out_size, void* d_ws, size_t ws_size,
                              hipStream_t stream) {
  (void)in_sizes; (void)n_in; (void)out_size; (void)ws_size;

  const float* x        = (const float*)d_in[0];
  const float* actor_h  = (const float*)d_in[1];
  const float* actor_c  = (const float*)d_in[2];
  const float* critic_h = (const float*)d_in[3];
  const float* critic_c = (const float*)d_in[4];
  const int*   done     = (const int*)d_in[5];
  const long long* action = (const long long*)d_in[6];

  const float* fw1[2]  = {(const float*)d_in[7],  (const float*)d_in[20]};
  const float* fb1[2]  = {(const float*)d_in[8],  (const float*)d_in[21]};
  const float* fw2[2]  = {(const float*)d_in[9],  (const float*)d_in[22]};
  const float* fb2[2]  = {(const float*)d_in[10], (const float*)d_in[23]};
  const float* flnw[2] = {(const float*)d_in[11], (const float*)d_in[24]};
  const float* flnb[2] = {(const float*)d_in[12], (const float*)d_in[25]};
  const float* wih[2]  = {(const float*)d_in[13], (const float*)d_in[26]};
  const float* whh[2]  = {(const float*)d_in[14], (const float*)d_in[27]};
  const float* lb[2]   = {(const float*)d_in[15], (const float*)d_in[28]};

  float* out = (float*)d_out;
  const size_t N = (size_t)Nn;

  // ---- workspace carve-out ----
  unsigned char* ws = (unsigned char*)d_ws;
  size_t off = 0;
  auto take = [&](size_t bytes) -> void* {
    void* ptr = ws + off;
    off = (off + bytes + 255) & ~(size_t)255;
    return ptr;
  };
  unsigned short *w1bf[2], *w2bf[2], *wihbf[2], *whhbf[2], *featbf[2];
  float *hsws[2], *hws[2], *cws[2];
  for (int br = 0; br < 2; ++br) {
    w1bf[br]  = (unsigned short*)take(F1d * OBS * 2);
    w2bf[br]  = (unsigned short*)take(Hd * F1d * 2);
    wihbf[br] = (unsigned short*)take(G4 * Hd * 2);
    whhbf[br] = (unsigned short*)take(G4 * Hd * 2);
  }
  for (int br = 0; br < 2; ++br) featbf[br] = (unsigned short*)take(N * Hd * 2);
  for (int br = 0; br < 2; ++br) hsws[br]   = (float*)take(N * Hd * 4);
  for (int br = 0; br < 2; ++br) { hws[br] = (float*)take((size_t)Bd * Hd * 4);
                                   cws[br] = (float*)take((size_t)Bd * Hd * 4); }

  // ---- 1. weights -> bf16 ----
  for (int br = 0; br < 2; ++br) {
    cvt_bf16_kernel<<<(F1d * OBS + 255) / 256, 256, 0, stream>>>(fw1[br], w1bf[br], F1d * OBS);
    cvt_bf16_kernel<<<(Hd * F1d + 255) / 256, 256, 0, stream>>>(fw2[br], w2bf[br], Hd * F1d);
    cvt_bf16_kernel<<<(G4 * Hd + 255) / 256, 256, 0, stream>>>(wih[br], wihbf[br], G4 * Hd);
    cvt_bf16_kernel<<<(G4 * Hd + 255) / 256, 256, 0, stream>>>(whh[br], whhbf[br], G4 * Hd);
  }

  // ---- 2. init h/c state ----
  {
    Copy4 p;
    p.s[0] = actor_h; p.s[1] = actor_c; p.s[2] = critic_h; p.s[3] = critic_c;
    p.d[0] = hws[0];  p.d[1] = cws[0];  p.d[2] = hws[1];   p.d[3] = cws[1];
    p.n = Bd * Hd;
    copy4_kernel<<<(Bd * Hd + 255) / 256, 256, 0, stream>>>(p);
  }

  // ---- 3. feature MLP (both branches) ----
  {
    FeatParams p;
    p.x = x;
    for (int br = 0; br < 2; ++br) {
      p.w1[br] = w1bf[br]; p.b1[br] = fb1[br];
      p.w2[br] = w2bf[br]; p.b2[br] = fb2[br];
      p.lnw[br] = flnw[br]; p.lnb[br] = flnb[br];
      p.feat[br] = featbf[br];
    }
    feat_kernel<<<dim3(Nn / 64, 2, 1), 256, 0, stream>>>(p);
  }

  // ---- 4. masked LSTM, T sequential steps ----
  {
    StepParams p;
    p.done = done;
    for (int br = 0; br < 2; ++br) {
      p.feat[br] = featbf[br]; p.wih[br] = wihbf[br]; p.whh[br] = whhbf[br];
      p.bias[br] = lb[br]; p.h[br] = hws[br]; p.c[br] = cws[br]; p.hs[br] = hsws[br];
    }
    for (int t = 0; t < Td; ++t) {
      p.t = t;
      lstm_step_kernel<<<dim3(Bd / 32, 1, 2), 256, 0, stream>>>(p);
    }
  }

  // ---- 5. actor head ----
  {
    ActorParams p;
    p.hs = hsws[0];
    p.lnw = (const float*)d_in[16]; p.lnb = (const float*)d_in[17];
    p.hw = (const float*)d_in[18];  p.hb = (const float*)d_in[19];
    p.action = action;
    p.out_action = out;          // [N]
    p.out_logp   = out + N;      // [N]
    p.out_ent    = out + 2 * N;  // [N]
    actor_head_kernel<<<Nn / 8, 256, 0, stream>>>(p);
  }

  // ---- 6. critic head ----
  critic_head_kernel<<<Nn / 8, 256, 0, stream>>>(
      hsws[1], (const float*)d_in[29], (const float*)d_in[30],
      (const float*)d_in[31], (const float*)d_in[32], out + 3 * N);

  // ---- 7. final h/c states ----
  {
    Copy4 p;
    p.s[0] = hws[0]; p.s[1] = cws[0]; p.s[2] = hws[1]; p.s[3] = cws[1];
    p.d[0] = out + 4 * N; p.d[1] = out + 5 * N;
    p.d[2] = out + 6 * N; p.d[3] = out + 7 * N;
    p.n = Bd * Hd;
    copy4_kernel<<<(Bd * Hd + 255) / 256, 256, 0, stream>>>(p);
  }
}
